// FastformerLayer_41532333752837
// MI455X (gfx1250) — matrix operand order repaired and checked by reading, not compiled, not run on hardware
//
#include <hip/hip_runtime.h>
#include <hip/hip_bf16.h>

// ---------------------------------------------------------------------------
// Fastformer layer for MI455X (gfx1250, wave32, WMMA + TDM).
// GEMMs (QKV, boom FFN, grouped causal conv taps) use v_wmma_f32_16x16x32_bf16
// with f32 accumulation. Weight (B) panels are DMA'd into LDS by the Tensor
// Data Mover (tensor_load_to_lds, TENSORcnt), double-buffered per K-step.
// ---------------------------------------------------------------------------

typedef __attribute__((ext_vector_type(16))) __bf16         v16bf;
typedef __attribute__((ext_vector_type(8)))  float          v8f;
typedef __attribute__((ext_vector_type(8)))  unsigned short u16x8;
typedef __attribute__((ext_vector_type(4)))  unsigned int   uint32x4;
typedef __attribute__((ext_vector_type(8)))  int            int32x8;
typedef __attribute__((ext_vector_type(4)))  int            int32x4;

#define BB   4
#define SS   2048
#define DD   1024
#define BSR  (BB * SS)     // 8192 rows
#define HHD  (4 * DD)      // 4096 boom width
#define GG   4
#define CPG  (DD / GG)     // 256 channels per conv group
#define KTAP 4

// TDM-staged B panel geometry for the generic GEMM.
#define BROWS   256        // N rows per block panel
#define KSTEP   64         // K elements per panel
#define BPADB   16         // LDS pad bytes inserted by TDM every 128B row
#define BSTRIDE (KSTEP * 2 + BPADB)   // 144B padded LDS row stride
#define BPANEL  (BROWS * BSTRIDE)     // 36864B per buffer

__device__ __forceinline__ unsigned short f2bf(float f) {
  unsigned int u = __float_as_uint(f);
  u += 0x7FFFu + ((u >> 16) & 1u);            // round-to-nearest-even
  return (unsigned short)(u >> 16);
}

__device__ __forceinline__ float gelu_exact(float x) {
  return 0.5f * x * (1.0f + erff(x * 0.7071067811865475f));
}

// Load one lane's 16 bf16 values for a WMMA A/B operand from two 16B chunks.
__device__ __forceinline__ v16bf ld_ab(const unsigned short* p0,
                                       const unsigned short* p1, bool valid) {
  union { u16x8 h[2]; v16bf v; } u;
  u16x8 z = {0, 0, 0, 0, 0, 0, 0, 0};
  u.h[0] = valid ? *(const u16x8*)p0 : z;
  u.h[1] = valid ? *(const u16x8*)p1 : z;
  return u.v;
}

// ------------------------- Tensor Data Mover -------------------------------
// DMA a [BROWS x KSTEP] bf16 tile (row stride K elements) from global memory
// into LDS at lds_off.  D# per CDNA5 ISA:
//   group0: count=1 | lds_addr | global_addr | type=2
//   group1: data_size=2B, pad_enable (16B pad per 128B row -> 144B LDS rows,
//           16B-aligned & bank-conflict-free), tensor dims/strides, tile dims
//   groups 2/3: zero (2-D tile).  Tracked with TENSORcnt.
__device__ __forceinline__ void tdm_load_b_panel(const unsigned short* gtile,
                                                 unsigned lds_off,
                                                 int K, int N) {
  unsigned long long ga = (unsigned long long)(uintptr_t)gtile;
  uint32x4 g0;
  g0[0] = 1u;                                        // count=1, user mode
  g0[1] = lds_off;                                   // LDS byte address
  g0[2] = (unsigned)(ga & 0xffffffffu);              // global_addr[31:0]
  g0[3] = (unsigned)((ga >> 32) & 0x1ffffffu)        // global_addr[56:32]
          | (2u << 30);                              // type=2 ("image")

  unsigned td0 = (unsigned)K;                        // tensor_dim0 (elements)
  unsigned td1 = (unsigned)N;                        // tensor_dim1 (rows)
  int32x8 g1;
  g1[0] = (1 << 16)                                  // data_size = 2 bytes
        | (1 << 20)                                  // pad_enable
        | (4 << 22)                                  // pad_interval: 32 dwords
        | (3 << 25);                                 // pad_amount: 4 dwords
  g1[1] = (int)((td0 & 0xffffu) << 16);              // tensor_dim0[15:0]
  g1[2] = (int)((td0 >> 16) | ((td1 & 0xffffu) << 16));
  g1[3] = (int)((td1 >> 16) | ((unsigned)KSTEP << 16));   // tile_dim0
  g1[4] = BROWS;                                     // tile_dim1 (tile_dim2=0)
  g1[5] = K;                                         // tensor_dim0_stride lo
  g1[6] = 0;                                         // stride hi / dim1_stride
  g1[7] = 0;
  int32x4 z4 = {0, 0, 0, 0};
  int32x8 z8 = {0, 0, 0, 0, 0, 0, 0, 0};
  // 6-arg form (clang-23 / therock HIP headers): (g0, g1, g2, g3, g4, cpol)
  __builtin_amdgcn_tensor_load_to_lds(g0, g1, z4, z4, z8, 0);
}

// --------------------------- weight packing --------------------------------
__global__ void cast_bf16_kernel(const float* __restrict__ src,
                                 unsigned short* __restrict__ dst, int n) {
  int i = blockIdx.x * 256 + threadIdx.x;
  if (i < n) dst[i] = f2bf(src[i]);
}

// conv_w [1024][256][4] (O,I,K) -> [4][1024][256] bf16 (tap-major, N=[O], K=[I])
__global__ void conv_pack_kernel(const float* __restrict__ w,
                                 unsigned short* __restrict__ dst) {
  int t = blockIdx.x * 256 + threadIdx.x;
  if (t < DD * CPG * KTAP) {
    int k = t & 3;
    int i = (t >> 2) & (CPG - 1);
    int o = t >> 10;
    dst[((size_t)k * DD + o) * CPG + i] = f2bf(w[t]);
  }
}

// ------------------------------ LayerNorm ----------------------------------
__global__ void ln_kernel(const float* __restrict__ x,
                          const float* __restrict__ g,
                          const float* __restrict__ b,
                          unsigned short* __restrict__ out) {
  int row  = blockIdx.x;
  int lane = threadIdx.x;                    // one wave32 per row
  const float* xr = x + (size_t)row * DD;
  float s = 0.f, s2 = 0.f;
  for (int i = lane; i < DD; i += 32) {
    float v = xr[i];
    s += v; s2 += v * v;
  }
  for (int o = 16; o > 0; o >>= 1) {
    s  += __shfl_xor(s,  o, 32);
    s2 += __shfl_xor(s2, o, 32);
  }
  float m   = s  * (1.0f / DD);
  float var = s2 * (1.0f / DD) - m * m;
  float inv = rsqrtf(var + 1e-5f);
  unsigned short* orow = out + (size_t)row * DD;
  for (int i = lane; i < DD; i += 32)
    orow[i] = f2bf((xr[i] - m) * inv * g[i] + b[i]);
}

// ------------------- grouped causal conv as 4 WMMA GEMM taps ---------------
// Block = 128 threads (4 waves). Block tile: M=16 rows, N=256 (one group).
__global__ void __launch_bounds__(128)
conv_gemm_kernel(const unsigned short* __restrict__ hln,
                 const unsigned short* __restrict__ wpk,
                 const float* __restrict__ convb,
                 const float* __restrict__ xin,
                 float* __restrict__ xout) {
  int lane   = threadIdx.x & 31;
  int wave   = threadIdx.x >> 5;
  int grp    = blockIdx.x;
  int nbase  = grp * CPG + wave * 64;
  int mbase  = blockIdx.y * 16;
  int laneLo = lane & 15;
  int laneHi = lane >> 4;

  v8f acc[4] = {};
  int m = mbase + laneLo;
  int s = m & (SS - 1);

#pragma unroll
  for (int tap = 0; tap < KTAP; ++tap) {
    int  shift = tap - (KTAP - 1);           // -3..0 (causal)
    bool valid = (s + shift) >= 0;
    const unsigned short* arow =
        hln + (size_t)(m + (valid ? shift : 0)) * DD + grp * CPG;
    const unsigned short* wb = wpk + (size_t)tap * DD * CPG;
    for (int kc = 0; kc < CPG; kc += 32) {
      v16bf a = ld_ab(arow + kc + laneHi * 16,
                      arow + kc + laneHi * 16 + 8, valid);
#pragma unroll
      for (int t = 0; t < 4; ++t) {
        const unsigned short* brow =
            wb + (size_t)(nbase + t * 16 + laneLo) * CPG + kc + laneHi * 16;
        v16bf bm = ld_ab(brow, brow + 8, true);
        acc[t] = __builtin_amdgcn_wmma_f32_16x16x32_bf16(
            false, a, false, bm, (short)0, acc[t], false, false);
      }
    }
  }

  int rowb = mbase + laneHi * 8;
#pragma unroll
  for (int t = 0; t < 4; ++t) {
    int col   = nbase + t * 16 + laneLo;
    float bia = convb[col];
#pragma unroll
    for (int vv = 0; vv < 8; ++vv) {
      size_t idx = (size_t)(rowb + vv) * DD + col;
      xout[idx]  = xin[idx] + bia + acc[t][vv];
    }
  }
}

// --------------------------- generic WMMA GEMM -----------------------------
// C[M,N] = A[M,K](bf16) * B[N,K](bf16)^T + bias.  B panels are TDM-staged in
// LDS (double-buffered, TENSORcnt-synchronized); A read direct from global.
//   EPI 0: f32 store (bias)            -> q/k/v projections
//   EPI 1: exact-GELU then bf16 store  -> boom up-projection
//   EPI 2: f32 store (bias + residual) -> boom down-projection (+skip)
// Block = 256 threads (8 waves, 2Mx4N). Block tile 32x256; wave tile 16x64.
template <int EPI>
__global__ void __launch_bounds__(256)
gemm_kernel(const unsigned short* __restrict__ A,
            const unsigned short* __restrict__ B,
            const float* __restrict__ bias,
            const float* __restrict__ resid,
            void* __restrict__ outp, int M, int N, int K) {
  __shared__ unsigned short bpan[2][BPANEL / 2];    // 2 x 36864B

  int lane   = threadIdx.x & 31;
  int wave   = threadIdx.x >> 5;
  int waveM  = wave >> 2;                    // 0..1
  int waveN  = wave & 3;                     // 0..3
  int mbase  = blockIdx.y * 32 + waveM * 16;
  int nblk   = blockIdx.x * 256;
  int laneLo = lane & 15;
  int laneHi = lane >> 4;

  const unsigned short* arow = A + (size_t)(mbase + laneLo) * K;
  const unsigned short* btile = B + (size_t)nblk * K;
  v8f acc[4] = {};

  if (wave == 0)
    tdm_load_b_panel(btile, (unsigned)(uintptr_t)&bpan[0][0], K, N);

  for (int ks = 0; ks < K; ks += KSTEP) {
    int cur = (ks / KSTEP) & 1;
    if (wave == 0) {
      if (ks + KSTEP < K) {
        tdm_load_b_panel(btile + (ks + KSTEP),
                         (unsigned)(uintptr_t)&bpan[cur ^ 1][0], K, N);
        __builtin_amdgcn_s_wait_tensorcnt(1);   // current panel landed
      } else {
        __builtin_amdgcn_s_wait_tensorcnt(0);
      }
    }
    __syncthreads();                            // panel visible to all waves

#pragma unroll
    for (int half = 0; half < 2; ++half) {
      int kc = ks + half * 32;
      v16bf a = ld_ab(arow + kc + laneHi * 16,
                      arow + kc + laneHi * 16 + 8, true);
#pragma unroll
      for (int t = 0; t < 4; ++t) {
        int nl = waveN * 64 + t * 16 + laneLo;  // padded 144B LDS rows
        const unsigned short* brow =
            bpan[cur] + nl * (BSTRIDE / 2) + half * 32 + laneHi * 16;
        v16bf bm = ld_ab(brow, brow + 8, true);
        acc[t] = __builtin_amdgcn_wmma_f32_16x16x32_bf16(
            false, a, false, bm, (short)0, acc[t], false, false);
      }
    }
    __syncthreads();                            // safe to overwrite buffer
  }

  int rowb = mbase + laneHi * 8;
#pragma unroll
  for (int t = 0; t < 4; ++t) {
    int col   = nblk + waveN * 64 + t * 16 + laneLo;
    float bia = bias[col];
#pragma unroll
    for (int vv = 0; vv < 8; ++vv) {
      size_t idx = (size_t)(rowb + vv) * N + col;
      float  r   = acc[t][vv] + bia;
      if constexpr (EPI == 0) {
        ((float*)outp)[idx] = r;
      } else if constexpr (EPI == 1) {
        ((unsigned short*)outp)[idx] = f2bf(gelu_exact(r));
      } else {
        ((float*)outp)[idx] = r + resid[idx];
      }
    }
  }
}

// ------------------- Fastformer additive-attention scan --------------------
__global__ void scan_kernel(const float* __restrict__ q,
                            const float* __restrict__ k,
                            const float* __restrict__ v,
                            float* __restrict__ x) {
  int d = blockIdx.x * blockDim.x + threadIdx.x;
  int b = blockIdx.y;
  size_t base = (size_t)b * SS * DD + d;
  float sq = 0.f, sk = 0.f;
  for (int s = 0; s < SS; ++s) {
    size_t i   = base + (size_t)s * DD;
    float  inv = 1.0f / (float)(s + 1);
    sq += q[i];
    float pq = sq * inv;
    sk += pq * k[i];
    float pk = sk * inv;
    x[i] += pk * v[i];
  }
}

// ---------------------------------------------------------------------------
extern "C" void kernel_launch(void* const* d_in, const int* in_sizes, int n_in,
                              void* d_out, int out_size, void* d_ws, size_t ws_size,
                              hipStream_t stream) {
  const float* x_in      = (const float*)d_in[0];
  const float* ln_attn_g = (const float*)d_in[2];
  const float* ln_attn_b = (const float*)d_in[3];
  const float* Wq = (const float*)d_in[4];   const float* bq = (const float*)d_in[5];
  const float* Wk = (const float*)d_in[6];   const float* bk = (const float*)d_in[7];
  const float* Wv = (const float*)d_in[8];   const float* bv = (const float*)d_in[9];
  const float* ln_conv_g = (const float*)d_in[10];
  const float* ln_conv_b = (const float*)d_in[11];
  const float* conv_w = (const float*)d_in[12];
  const float* conv_b = (const float*)d_in[13];
  const float* ln_boom_g = (const float*)d_in[14];
  const float* ln_boom_b = (const float*)d_in[15];
  const float* Wb = (const float*)d_in[16];  const float* bb = (const float*)d_in[17];
  const float* Wu = (const float*)d_in[18];  const float* bu = (const float*)d_in[19];
  float* out = (float*)d_out;

  // ---- workspace carve-up (total 168 MB) ----
  char* ws = (char*)d_ws;
  size_t off = 0;
  unsigned short* hln = (unsigned short*)(ws + off); off += (size_t)BSR * DD * 2;
  unsigned short* wqb = (unsigned short*)(ws + off); off += (size_t)DD * DD * 2;
  unsigned short* wkb = (unsigned short*)(ws + off); off += (size_t)DD * DD * 2;
  unsigned short* wvb = (unsigned short*)(ws + off); off += (size_t)DD * DD * 2;
  unsigned short* wbb = (unsigned short*)(ws + off); off += (size_t)HHD * DD * 2;
  unsigned short* wub = (unsigned short*)(ws + off); off += (size_t)DD * HHD * 2;
  unsigned short* wcp = (unsigned short*)(ws + off); off += (size_t)KTAP * DD * CPG * 2;
  float* x1 = (float*)(ws + off); off += (size_t)BSR * DD * 4;  // x1, then x2 in-place
  float* q  = (float*)(ws + off); off += (size_t)BSR * DD * 4;
  float* kk = (float*)(ws + off); off += (size_t)BSR * DD * 4;
  float* vv = (float*)(ws + off); off += (size_t)BSR * DD * 4;
  unsigned short* hb = (unsigned short*)q;   // boom intermediate reuses q+k

  // ---- 0) pack weights to bf16 (B layout = [N][K], K contiguous) ----
  {
    int n1 = DD * DD, n2 = HHD * DD, nc = DD * CPG * KTAP;
    cast_bf16_kernel<<<(n1 + 255) / 256, 256, 0, stream>>>(Wq, wqb, n1);
    cast_bf16_kernel<<<(n1 + 255) / 256, 256, 0, stream>>>(Wk, wkb, n1);
    cast_bf16_kernel<<<(n1 + 255) / 256, 256, 0, stream>>>(Wv, wvb, n1);
    cast_bf16_kernel<<<(n2 + 255) / 256, 256, 0, stream>>>(Wb, wbb, n2);
    cast_bf16_kernel<<<(n2 + 255) / 256, 256, 0, stream>>>(Wu, wub, n2);
    conv_pack_kernel<<<(nc + 255) / 256, 256, 0, stream>>>(conv_w, wcp);
  }

  // ---- 1) conv stem: x1 = x + conv(LN_conv(x)) ----
  ln_kernel<<<BSR, 32, 0, stream>>>(x_in, ln_conv_g, ln_conv_b, hln);
  conv_gemm_kernel<<<dim3(GG, BSR / 16), 128, 0, stream>>>(hln, wcp, conv_b, x_in, x1);

  // ---- 2) Fastformer attention: q/k/v projections + cumulative scan ----
  ln_kernel<<<BSR, 32, 0, stream>>>(x1, ln_attn_g, ln_attn_b, hln);
  dim3 gqkv(DD / 256, BSR / 32);
  gemm_kernel<0><<<gqkv, 256, 0, stream>>>(hln, wqb, bq, nullptr, q,  BSR, DD, DD);
  gemm_kernel<0><<<gqkv, 256, 0, stream>>>(hln, wkb, bk, nullptr, kk, BSR, DD, DD);
  gemm_kernel<0><<<gqkv, 256, 0, stream>>>(hln, wvb, bv, nullptr, vv, BSR, DD, DD);
  scan_kernel<<<dim3(DD / 256, BB), 256, 0, stream>>>(q, kk, vv, x1);

  // ---- 3) boom FFN: out = x2 + Wu(gelu(Wb(LN_boom(x2))+bb))+bu ----
  ln_kernel<<<BSR, 32, 0, stream>>>(x1, ln_boom_g, ln_boom_b, hln);
  gemm_kernel<1><<<dim3(HHD / 256, BSR / 32), 256, 0, stream>>>(hln, wbb, bb, nullptr, hb, BSR, HHD, DD);
  gemm_kernel<2><<<dim3(DD / 256, BSR / 32), 256, 0, stream>>>(hb, wub, bu, x1, out, BSR, DD, HHD);

  (void)in_sizes; (void)n_in; (void)out_size; (void)ws_size;
}
